// MultiHeadAttention_44890998178552
// MI455X (gfx1250) — matrix operand order repaired and checked by reading, not compiled, not run on hardware
//
#include <hip/hip_runtime.h>
#include <hip/hip_bf16.h>
#include <stdint.h>

typedef __attribute__((ext_vector_type(16))) __bf16 v16bf;
typedef __attribute__((ext_vector_type(2)))  __bf16 v2bf;
typedef __attribute__((ext_vector_type(8)))  float  v8f;

#define B_   4
#define S_   2048
#define DM   1024
#define H_   16
#define DK   64
#define MTOT (B_ * S_)

// Pack two fp32 -> one dword of 2x bf16.
// Hardware packed convert when available, else bias + one v_perm_b32
// (round-half-up; inputs finite, <=1ulp-of-bf16 difference vs RNE).
__device__ __forceinline__ uint32_t pk_bf16(float a, float b) {
#if __has_builtin(__builtin_amdgcn_cvt_pk_bf16_f32)
    return __builtin_bit_cast(uint32_t, __builtin_amdgcn_cvt_pk_bf16_f32(a, b));
#else
    uint32_t ua = __builtin_bit_cast(uint32_t, a) + 0x8000u;
    uint32_t ub = __builtin_bit_cast(uint32_t, b) + 0x8000u;
    return __builtin_amdgcn_perm(ub, ua, 0x07060302u);
#endif
}

// compare-select max: avoids v_max canonicalize (inputs are finite)
__device__ __forceinline__ float fmax_fast(float a, float b) { return a > b ? a : b; }

__device__ __forceinline__ v2bf pkmax(v2bf a, v2bf b) {
    return __builtin_elementwise_max(a, b);        // v_pk_max_num_bf16
}

union FragU { uint4 u[2]; v16bf v; };

// Load a 16-element bf16 fragment as two 16-byte chunks (b128 loads).
__device__ __forceinline__ v16bf ld_frag(const __bf16* p0, const __bf16* p1) {
    FragU f;
    f.u[0] = *(const uint4*)p0;
    f.u[1] = *(const uint4*)p1;
    return f.v;
}

// Packed bf16 butterfly max over 16-lane groups (1 DS + 1 pk-max per step)
#define SWZ_PKMAX(IMM)                                                         \
    _Pragma("unroll")                                                          \
    for (int jj = 0; jj < 4; ++jj) {                                           \
        pm[jj] = pkmax(pm[jj], __builtin_bit_cast(v2bf,                        \
            __builtin_amdgcn_ds_swizzle(__builtin_bit_cast(int, pm[jj]), (IMM)))); \
    }

// ---------------------------------------------------------------------------
// fp32 -> bf16 weight conversion (vectorized, packed converts, b64 stores)
// ---------------------------------------------------------------------------
__global__ __launch_bounds__(256) void cvt_f32_bf16(const float* __restrict__ src,
                                                    __bf16* __restrict__ dst, int n4) {
    int i = blockIdx.x * 256 + threadIdx.x;
    if (i >= n4) return;
    float4 v = ((const float4*)src)[i];
    uint2 w;
    w.x = pk_bf16(v.x, v.y);
    w.y = pk_bf16(v.z, v.w);
    *(uint2*)(dst + (size_t)i * 4) = w;
}

// ---------------------------------------------------------------------------
// Tiled WMMA GEMM: C[M,N] = A[M,1024] x B[1024,1024]
//   A_FP32:    A fp32, packed-converted to bf16 while staging into LDS.
//              A bf16:  staged with async global->LDS DMA (ASYNCcnt).
//   HEADSPLIT: output bf16 in [b,h,s,64] layout (with scale); else fp32 [M,1024]
// 256 threads (8 waves), block tile 256x64, K-step 64 (two 32-K sub-steps).
// Wave grid 4(M)x2(N); wave tile 64x32 -> 16 WMMA per barrier pair.
// ---------------------------------------------------------------------------
template<bool A_FP32, bool HEADSPLIT>
__global__ __launch_bounds__(256) void gemm_wmma(const void* __restrict__ Ap,
                                                 const __bf16* __restrict__ Bw,
                                                 void* __restrict__ outp, float scale) {
    constexpr int BM = 256, BN = 64, BK = 64, LDP = BK + 8;
    // single flat LDS symbol => frame offset 0; sA at byte 0, sBt after it
    __shared__ __bf16 smem[BM * LDP + BN * LDP];   // 36 KB + 9 KB
    __bf16* sA  = smem;                // A tile (bf16), row pitch LDP
    __bf16* sBt = smem + BM * LDP;     // B tile transposed [n][k]

    const int tid  = threadIdx.x;
    const int lane = tid & 31;
    const int wid  = tid >> 5;
    const int waveM = wid >> 1, waveN = wid & 1;
    const int m0 = blockIdx.x * BM, n0 = blockIdx.y * BN;
    const int lr = lane & 15;
    const int hA = (lane < 16) ? 0 : 8;    // A-frag K-group base (and C row offset)
    const int kB = (lane < 16) ? 0 : 16;   // unused; B frags use hA, same k order as A

    v8f acc[4][2] = {};

    for (int k0 = 0; k0 < DM; k0 += BK) {
        __syncthreads();
        if constexpr (A_FP32) {
            const float* Af = (const float*)Ap;
            #pragma unroll
            for (int p = 0; p < 16; ++p) {
                int flat = p * 1024 + tid * 4;
                int row = flat >> 6, col = flat & 63;
                const float* gp = Af + (size_t)(m0 + row) * DM + k0 + col;
                float4 v = *(const float4*)gp;
                if (p == 0 && k0 + BK < DM) __builtin_prefetch(gp + BK, 0, 1);
                uint2 w;
                w.x = pk_bf16(v.x, v.y);
                w.y = pk_bf16(v.z, v.w);
                *(uint2*)&sA[row * LDP + col] = w;
            }
        } else {
            // Async DMA: per-lane 16B global->LDS copies, tracked by ASYNCcnt.
            const __bf16* Ab = (const __bf16*)Ap;
            #pragma unroll
            for (int p = 0; p < 8; ++p) {
                int flat = p * 2048 + tid * 8;
                int row = flat >> 6, col = flat & 63;
                const __bf16* gp = Ab + (size_t)(m0 + row) * DM + k0 + col;
                uint32_t ldsoff = (uint32_t)(2 * (row * LDP + col));  // sA at LDS offset 0
                asm volatile("global_load_async_to_lds_b128 %0, %1, off"
                             :: "v"(ldsoff), "v"(gp) : "memory");
            }
            asm volatile("s_wait_asynccnt 0x0" ::: "memory");
        }
        {   // B tile: 64x64, load coalesced rows, store transposed into LDS
            #pragma unroll
            for (int p = 0; p < 2; ++p) {
                int k = p * 32 + (tid >> 3), cg = (tid & 7) * 8;
                const __bf16* gp = Bw + (size_t)(k0 + k) * DM + n0 + cg;
                uint4 v = *(const uint4*)gp;
                if (p == 0 && k0 + BK < DM) __builtin_prefetch(gp + (size_t)BK * DM, 0, 1);
                const __bf16* e = (const __bf16*)&v;
                #pragma unroll
                for (int j = 0; j < 8; ++j) sBt[(cg + j) * LDP + k] = e[j];
            }
        }
        __syncthreads();

        #pragma unroll
        for (int kk = 0; kk < 2; ++kk) {
            const int ko = kk * 32;
            v16bf a[4], b[2];
            #pragma unroll
            for (int mi = 0; mi < 4; ++mi) {
                const int r = waveM * 64 + mi * 16 + lr;
                a[mi] = ld_frag(&sA[r * LDP + ko + hA], &sA[r * LDP + ko + hA + 16]);
            }
            #pragma unroll
            for (int ni = 0; ni < 2; ++ni) {
                const int n = waveN * 32 + ni * 16 + lr;
                b[ni] = ld_frag(&sBt[n * LDP + ko + hA], &sBt[n * LDP + ko + hA + 16]);
            }
            #pragma unroll
            for (int mi = 0; mi < 4; ++mi)
                #pragma unroll
                for (int ni = 0; ni < 2; ++ni)
                    acc[mi][ni] = __builtin_amdgcn_wmma_f32_16x16x32_bf16(
                        false, a[mi], false, b[ni], (short)0, acc[mi][ni], false, false);
        }
    }

    // Epilogue: pair rows (2jj, 2jj+1) -> packed convert, b16 + d16_hi stores
    #pragma unroll
    for (int mi = 0; mi < 4; ++mi)
        #pragma unroll
        for (int ni = 0; ni < 2; ++ni)
            #pragma unroll
            for (int jj = 0; jj < 4; ++jj) {
                float v0 = acc[mi][ni][2 * jj]     * scale;
                float v1 = acc[mi][ni][2 * jj + 1] * scale;
                int row = m0 + waveM * 64 + mi * 16 + 2 * jj + hA;
                int col = n0 + waveN * 32 + ni * 16 + lr;
                if constexpr (HEADSPLIT) {
                    uint32_t w = pk_bf16(v0, v1);
                    int bb = row >> 11, sr = row & (S_ - 1);
                    int h = col >> 6, d = col & (DK - 1);
                    __bf16* p = (__bf16*)outp + (((size_t)(bb * H_ + h)) * S_ + sr) * DK + d;
                    p[0]  = __builtin_bit_cast(__bf16, (uint16_t)w);
                    p[DK] = __builtin_bit_cast(__bf16, (uint16_t)(w >> 16));
                } else {
                    float* p = (float*)outp + (size_t)row * DM + col;
                    p[0]  = v0;
                    p[DM] = v1;
                }
            }
}

// ---------------------------------------------------------------------------
// Flash attention: O = softmax(Q K^T) V per (b,h). Q pre-scaled by 1/sqrt(dk).
// Grid (b*h, s/128). 8 waves x 16 query rows. All matmuls via bf16 WMMA.
// P columns stored permuted (key tc*16+n -> column 2n+tc) so (p0,p1) pairs
// pack into one b32 store; V rows staged into LDS in the same permuted order.
// Row max: packed-bf16 ds_swizzle butterfly (safe softmax shift).
// Row sums: ones-matrix WMMA. State updates vectorized (v_pk_* f32).
// ---------------------------------------------------------------------------
__global__ __launch_bounds__(256) void flash_fwd(const __bf16* __restrict__ Q,
                                                 const __bf16* __restrict__ K,
                                                 const __bf16* __restrict__ V,
                                                 __bf16* __restrict__ ctx) {
    __shared__ __bf16 sVt[DK][32 + 8];   // V tile transposed+permuted: [d][pos]
    __shared__ __bf16 sP[8][16][32];     // per-wave P tile (C-layout -> A-layout relay)

    const int bh = blockIdx.x;
    const int rb = blockIdx.y;
    const size_t base = (size_t)bh * S_ * DK;
    const __bf16* Qp = Q + base;
    const __bf16* Kp = K + base;
    const __bf16* Vp = V + base;

    const int tid  = threadIdx.x;
    const int lane = tid & 31;
    const int wid  = tid >> 5;
    const int lr    = lane & 15;
    const int hlo   = (lane < 16) ? 0 : 8;
    const int khalf = (lane < 16) ? 0 : 16;
    const int rowbase = rb * 128 + wid * 16;

    // Q fragments: 16 rows x 64 d, as two K=32 chunks (A-layout)
    v16bf qf[2];
    #pragma unroll
    for (int c = 0; c < 2; ++c) {
        const __bf16* p = Qp + (size_t)(rowbase + lr) * DK + c * 32 + hlo;
        qf[c] = ld_frag(p, p + 16);
    }

    // all-ones B fragment for row-sum-by-WMMA (dword immediates, no inserts)
    FragU onesU;
    onesU.u[0] = uint4{0x3F803F80u, 0x3F803F80u, 0x3F803F80u, 0x3F803F80u};
    onesU.u[1] = onesU.u[0];
    const v16bf ones = onesU.v;

    v8f accO[4] = {};
    v8f l_v = {};
    float m_[8];
    #pragma unroll
    for (int j = 0; j < 8; ++j) m_[j] = -1e30f;

    for (int t0 = 0; t0 < S_; t0 += 32) {
        __syncthreads();
        {   // stage V tile (32 x 64) transposed into LDS, key order permuted
            int k = tid >> 3, cg = (tid & 7) * 8;
            int pos = ((k & 15) << 1) | (k >> 4);   // key tc*16+n -> 2n+tc
            uint4 v = *(const uint4*)(Vp + (size_t)(t0 + k) * DK + cg);
            const __bf16* e = (const __bf16*)&v;
            #pragma unroll
            for (int j = 0; j < 8; ++j) sVt[cg + j][pos] = e[j];
        }
        __syncthreads();

        // S = Q K^T for two 16-column sub-tiles; K rows are B-fragment friendly
        v8f s01[2];
        #pragma unroll
        for (int tc = 0; tc < 2; ++tc) {
            const int r = t0 + tc * 16 + lr;
            v8f s = {};
            #pragma unroll
            for (int c = 0; c < 2; ++c) {
                const __bf16* p = Kp + (size_t)r * DK + c * 32 + hlo;
                v16bf kb = ld_frag(p, p + 16);
                s = __builtin_amdgcn_wmma_f32_16x16x32_bf16(false, qf[c], false, kb, (short)0, s, false, false);
            }
            s01[tc] = s;
        }

        // Row-max via packed-bf16 swizzle butterfly (rows 2jj,2jj+1 per dword)
        v2bf pm[4];
        #pragma unroll
        for (int jj = 0; jj < 4; ++jj) {
            float c0 = fmax_fast(s01[0][2 * jj],     s01[1][2 * jj]);
            float c1 = fmax_fast(s01[0][2 * jj + 1], s01[1][2 * jj + 1]);
            pm[jj] = __builtin_bit_cast(v2bf, pk_bf16(c0, c1));
        }
        SWZ_PKMAX(0x201F);   // xor 8
        SWZ_PKMAX(0x101F);   // xor 4
        SWZ_PKMAX(0x081F);   // xor 2
        SWZ_PKMAX(0x041F);   // xor 1

        // Online softmax: bf16-rounded max is a valid (safe) shift
        v8f fv;
        #pragma unroll
        for (int j = 0; j < 8; ++j) {
            float vmax = (float)pm[j >> 1][j & 1];
            float nm = fmax_fast(m_[j], vmax);
            fv[j] = __expf(m_[j] - nm);
            m_[j] = nm;
            float p0 = __expf(s01[0][j] - nm);
            float p1 = __expf(s01[1][j] - nm);
            *(uint32_t*)&sP[wid][j + hlo][2 * lr] = pk_bf16(p0, p1);
        }
        // intra-wave LDS RAW: wait for our own DS stores before fragment reload
        asm volatile("s_wait_dscnt 0x0" ::: "memory");

        v16bf pA = ld_frag(&sP[wid][lr][hlo], &sP[wid][lr][hlo + 16]);

        // row sums of P via WMMA against all-ones (replicated across row lanes)
        v8f zero = {};
        v8f rsum = __builtin_amdgcn_wmma_f32_16x16x32_bf16(false, pA, false, ones, (short)0, zero, false, false);

        l_v = l_v * fv + rsum;            // v_pk_fma_f32
        #pragma unroll
        for (int nd = 0; nd < 4; ++nd) accO[nd] *= fv;   // v_pk_mul_f32

        #pragma unroll
        for (int nd = 0; nd < 4; ++nd) {
            v16bf vb = ld_frag(&sVt[nd * 16 + lr][hlo], &sVt[nd * 16 + lr][hlo + 16]);
            accO[nd] = __builtin_amdgcn_wmma_f32_16x16x32_bf16(false, pA, false, vb, (short)0, accO[nd], false, false);
        }
    }

    // Epilogue: reciprocal once per row, packed converts, b16 + d16_hi stores
    v8f rv;
    #pragma unroll
    for (int j = 0; j < 8; ++j) rv[j] = __builtin_amdgcn_rcpf(l_v[j]);

    const int bb = bh >> 4, h = bh & 15;
    #pragma unroll
    for (int nd = 0; nd < 4; ++nd) {
        v8f o = accO[nd] * rv;            // v_pk_mul_f32
        #pragma unroll
        for (int jj = 0; jj < 4; ++jj) {
            uint32_t w = pk_bf16(o[2 * jj], o[2 * jj + 1]);
            int sr = rowbase + 2 * jj + hlo;
            __bf16* p = ctx + ((size_t)(bb * S_ + sr)) * DM + h * DK + nd * 16 + lr;
            p[0]  = __builtin_bit_cast(__bf16, (uint16_t)w);
            p[DM] = __builtin_bit_cast(__bf16, (uint16_t)(w >> 16));
        }
    }
}

// ---------------------------------------------------------------------------
extern "C" void kernel_launch(void* const* d_in, const int* in_sizes, int n_in,
                              void* d_out, int out_size, void* d_ws, size_t ws_size,
                              hipStream_t stream) {
    const float* Xq = (const float*)d_in[0];
    const float* Xk = (const float*)d_in[1];
    const float* Xv = (const float*)d_in[2];
    const float* Wq = (const float*)d_in[3];
    const float* Wk = (const float*)d_in[4];
    const float* Wv = (const float*)d_in[5];
    const float* Wo = (const float*)d_in[6];

    char* ws = (char*)d_ws;
    const size_t WSZ = (size_t)DM * DM * sizeof(__bf16);           // 2 MB per weight
    const size_t QSZ = (size_t)B_ * H_ * S_ * DK * sizeof(__bf16); // 16 MB per tensor
    __bf16* Wqb = (__bf16*)(ws + 0 * WSZ);
    __bf16* Wkb = (__bf16*)(ws + 1 * WSZ);
    __bf16* Wvb = (__bf16*)(ws + 2 * WSZ);
    __bf16* Wob = (__bf16*)(ws + 3 * WSZ);
    __bf16* Qb  = (__bf16*)(ws + 4 * WSZ);
    __bf16* Kb  = (__bf16*)(ws + 4 * WSZ + 1 * QSZ);
    __bf16* Vb  = (__bf16*)(ws + 4 * WSZ + 2 * QSZ);
    __bf16* Cb  = (__bf16*)(ws + 4 * WSZ + 3 * QSZ);               // ctx [b,s,1024]

    const int n4 = DM * DM / 4;
    dim3 cgrid((n4 + 255) / 256);
    cvt_f32_bf16<<<cgrid, 256, 0, stream>>>(Wq, Wqb, n4);
    cvt_f32_bf16<<<cgrid, 256, 0, stream>>>(Wk, Wkb, n4);
    cvt_f32_bf16<<<cgrid, 256, 0, stream>>>(Wv, Wvb, n4);
    cvt_f32_bf16<<<cgrid, 256, 0, stream>>>(Wo, Wob, n4);

    dim3 ggrid(MTOT / 256, DM / 64);  // (32, 16)
    // fold attention scale 1/sqrt(64) into Q
    gemm_wmma<true, true><<<ggrid, 256, 0, stream>>>(Xq, Wqb, Qb, 0.125f);
    gemm_wmma<true, true><<<ggrid, 256, 0, stream>>>(Xk, Wkb, Kb, 1.0f);
    gemm_wmma<true, true><<<ggrid, 256, 0, stream>>>(Xv, Wvb, Vb, 1.0f);

    flash_fwd<<<dim3(B_ * H_, S_ / 128), 256, 0, stream>>>(Qb, Kb, Vb, Cb);

    gemm_wmma<false, false><<<ggrid, 256, 0, stream>>>(Cb, Wob, d_out, 1.0f);
}
